// SilhouetteLoss_69741678953083
// MI455X (gfx1250) — matrix operand order repaired and checked by reading, not compiled, not run on hardware
//
#include <hip/hip_runtime.h>
#include <hip/hip_bf16.h>

typedef __attribute__((ext_vector_type(16))) __bf16 v16bf;
typedef __attribute__((ext_vector_type(8)))  float  v8f;

#define N_PTS   16384
#define NDIM    64
#define NCLUST  10
#define COL_CHUNKS 16

struct Frag32B { uint4 a, b; };

static __device__ __forceinline__ v16bf make_frag(uint4 x, uint4 y) {
  Frag32B f{x, y};
  return __builtin_bit_cast(v16bf, f);
}

// ---------------- pass 0: zero accumulators, init output to 1.0 ----------------
__global__ void zero_kernel(float* __restrict__ S, int* __restrict__ cnt, float* __restrict__ out) {
  int i = blockIdx.x * blockDim.x + threadIdx.x;
  if (i < N_PTS * NCLUST) S[i] = 0.0f;
  if (i < NCLUST) cnt[i] = 0;
  if (i == 0) out[0] = 1.0f;
}

// ---------------- pass 1: bf16 hi/lo split, squared norms, cluster counts ------
__global__ void prep_kernel(const float* __restrict__ X, const int* __restrict__ labels,
                            __bf16* __restrict__ fh, __bf16* __restrict__ fl,
                            float* __restrict__ sq, int* __restrict__ cnt) {
  int i = blockIdx.x * blockDim.x + threadIdx.x;
  if (i >= N_PTS) return;
  const float* row = X + (size_t)i * NDIM;
  float s = 0.0f;
  #pragma unroll 8
  for (int d = 0; d < NDIM; ++d) {
    float x = row[d];
    s += x * x;
    __bf16 h = (__bf16)x;
    float r = x - (float)h;
    fh[(size_t)i * NDIM + d] = h;
    fl[(size_t)i * NDIM + d] = (__bf16)r;
  }
  sq[i] = s;
  atomicAdd(&cnt[labels[i]], 1);
}

// ---------------- pass 2: fused WMMA distance tiles + segmented sums -----------
__global__ __launch_bounds__(32)
void tile_kernel(const __bf16* __restrict__ fh, const __bf16* __restrict__ fl,
                 const float* __restrict__ sq, const int* __restrict__ labels,
                 float* __restrict__ S) {
  __shared__ float Slds[16 * NCLUST];
  const int lane = threadIdx.x;
  for (int i = lane; i < 16 * NCLUST; i += 32) Slds[i] = 0.0f;

  const int I    = blockIdx.x * 16;      // row tile base
  const int half = lane >> 4;            // 0: lanes 0-15, 1: lanes 16-31
  const int lm   = lane & 15;

  // A fragments for rows I..I+15, hoisted over the whole column sweep.
  // 16-bit A 16x32 layout: lane<16 -> K {0..7,16..23}; lane>=16 -> K {8..15,24..31}.
  const __bf16* ah = fh + (size_t)(I + lm) * NDIM;
  const __bf16* al = fl + (size_t)(I + lm) * NDIM;
  v16bf Ah0 = make_frag(*(const uint4*)(ah +  0 + half * 8), *(const uint4*)(ah + 16 + half * 8));
  v16bf Ah1 = make_frag(*(const uint4*)(ah + 32 + half * 8), *(const uint4*)(ah + 48 + half * 8));
  v16bf Al0 = make_frag(*(const uint4*)(al +  0 + half * 8), *(const uint4*)(al + 16 + half * 8));
  v16bf Al1 = make_frag(*(const uint4*)(al + 32 + half * 8), *(const uint4*)(al + 48 + half * 8));

  float sqr[8];
  #pragma unroll
  for (int v = 0; v < 8; ++v) sqr[v] = sq[I + half * 8 + v];

  const int tilesPerChunk = (N_PTS / 16) / COL_CHUNKS;
  const int J0 = blockIdx.y * tilesPerChunk * 16;

  // Unroll by 2: two independent WMMA chains + epilogues let the scheduler
  // fill the WMMA->VALU RAW hazard slots with the other tile's work.
  #pragma unroll 2
  for (int t = 0; t < tilesPerChunk; ++t) {
    const int col = J0 + t * 16 + lm;
    const __bf16* bh = fh + (size_t)col * NDIM;
    const __bf16* bl = fl + (size_t)col * NDIM;
    // 16-bit B 32x16 layout: lane<16 -> K {0..7,16..23} of column N=lane; lane>=16 -> K {8..15,24..31}.
    v16bf Bh0 = make_frag(*(const uint4*)(bh +  0 + half * 8), *(const uint4*)(bh + 16 + half * 8));
    v16bf Bh1 = make_frag(*(const uint4*)(bh + 32 + half * 8), *(const uint4*)(bh + 48 + half * 8));
    v16bf Bl0 = make_frag(*(const uint4*)(bl +  0 + half * 8), *(const uint4*)(bl + 16 + half * 8));
    v16bf Bl1 = make_frag(*(const uint4*)(bl + 32 + half * 8), *(const uint4*)(bl + 48 + half * 8));

    // gram ≈ h·h + h·l + l·h  (near-fp32 dot via bf16 split; l·l ~ 2^-16 dropped)
    v8f acc = {};
    acc = __builtin_amdgcn_wmma_f32_16x16x32_bf16(false, Ah0, false, Bh0, (short)0, acc, false, false);
    acc = __builtin_amdgcn_wmma_f32_16x16x32_bf16(false, Ah1, false, Bh1, (short)0, acc, false, false);
    acc = __builtin_amdgcn_wmma_f32_16x16x32_bf16(false, Ah0, false, Bl0, (short)0, acc, false, false);
    acc = __builtin_amdgcn_wmma_f32_16x16x32_bf16(false, Ah1, false, Bl1, (short)0, acc, false, false);
    acc = __builtin_amdgcn_wmma_f32_16x16x32_bf16(false, Al0, false, Bh0, (short)0, acc, false, false);
    acc = __builtin_amdgcn_wmma_f32_16x16x32_bf16(false, Al1, false, Bh1, (short)0, acc, false, false);

    const float sqc = sq[col];
    const int   lab = labels[col];
    #pragma unroll
    for (int v = 0; v < 8; ++v) {
      float sqd  = sqr[v] + sqc - 2.0f * acc[v];
      // raw v_sqrt_f32: operands are well-scaled squared distances, no range fixup needed
      float dist = sqd > 0.0f ? __builtin_amdgcn_sqrtf(sqd) : 0.0f;
      atomicAdd(&Slds[(half * 8 + v) * NCLUST + lab], dist);   // ds_add_f32
    }
  }

  __syncthreads();
  for (int i = lane; i < 16 * NCLUST; i += 32)
    atomicAdd(&S[(size_t)(I + i / NCLUST) * NCLUST + (i % NCLUST)], Slds[i]);
}

// ---------------- pass 3: per-point silhouette, mean, out = 1 - mean -----------
__global__ void finalize_kernel(const float* __restrict__ S, const int* __restrict__ labels,
                                const int* __restrict__ cnt, float* __restrict__ out) {
  __shared__ float red[256];
  int i = blockIdx.x * blockDim.x + threadIdx.x;
  float score = 0.0f;
  if (i < N_PTS) {
    int   lab   = labels[i];
    float ci    = (float)cnt[lab];
    float intra = S[(size_t)i * NCLUST + lab] / fmaxf(ci - 1.0f, 1.0f);
    float minv  = INFINITY;
    #pragma unroll
    for (int k = 0; k < NCLUST; ++k) {
      if (k == lab || cnt[k] == 0) continue;
      minv = fminf(minv, S[(size_t)i * NCLUST + k] / (float)cnt[k]);
    }
    float denom = fmaxf(minv, intra);
    if (ci > 1.0f && denom > 0.0f) score = (minv - intra) / denom;
  }
  red[threadIdx.x] = score;
  __syncthreads();
  for (int s = 128; s > 0; s >>= 1) {
    if (threadIdx.x < s) red[threadIdx.x] += red[threadIdx.x + s];
    __syncthreads();
  }
  if (threadIdx.x == 0) atomicAdd(out, -red[0] / (float)N_PTS);
}

extern "C" void kernel_launch(void* const* d_in, const int* in_sizes, int n_in,
                              void* d_out, int out_size, void* d_ws, size_t ws_size,
                              hipStream_t stream) {
  (void)in_sizes; (void)n_in; (void)out_size; (void)ws_size;
  const float* X      = (const float*)d_in[0];
  const int*   labels = (const int*)d_in[1];
  float*       out    = (float*)d_out;

  char* ws = (char*)d_ws;
  float*  S   = (float*)ws;   ws += (size_t)N_PTS * NCLUST * sizeof(float);   // 640 KB
  int*    cnt = (int*)ws;     ws += 256;                                      // padded
  float*  sq  = (float*)ws;   ws += (size_t)N_PTS * sizeof(float);            // 64 KB
  __bf16* fh  = (__bf16*)ws;  ws += (size_t)N_PTS * NDIM * sizeof(__bf16);    // 2 MB
  __bf16* fl  = (__bf16*)ws;                                                  // 2 MB

  zero_kernel<<<(N_PTS * NCLUST + 255) / 256, 256, 0, stream>>>(S, cnt, out);
  prep_kernel<<<(N_PTS + 255) / 256, 256, 0, stream>>>(X, labels, fh, fl, sq, cnt);
  dim3 grid(N_PTS / 16, COL_CHUNKS);
  tile_kernel<<<grid, 32, 0, stream>>>(fh, fl, sq, labels, S);
  finalize_kernel<<<(N_PTS + 255) / 256, 256, 0, stream>>>(S, labels, cnt, out);
}
